// NetVLAD_1941325218409
// MI455X (gfx1250) — matrix operand order repaired and checked by reading, not compiled, not run on hardware
//
#include <hip/hip_runtime.h>
#include <hip/hip_bf16.h>

// ---------------------------------------------------------------------------
// NetVLAD fused forward for MI455X (gfx1250, wave32, WMMA bf16 16x16x32)
//
// Shapes: N=64 images, D=256 channels, T=H*W=4096 pixels, K=64 clusters.
// GEMM1: logits = conv_w(K,D) @ x(D,T)   -> bf16 WMMA, fp32 acc
// softmax over K, fold 1/||x_t|| into logits pre-softmax and into a' post.
// GEMM2: vlad  += a'(K,T) @ x^T(T,D)     -> bf16 WMMA, fp32 acc
// x read from HBM exactly once (256 MB), both GEMMs fed from LDS.
// ---------------------------------------------------------------------------

typedef __attribute__((ext_vector_type(16))) __bf16 v16bf;
typedef __attribute__((ext_vector_type(8)))  __bf16 v8bf;
typedef __attribute__((ext_vector_type(4)))  __bf16 v4bf;
typedef __attribute__((ext_vector_type(8)))  float  v8f;

#define N_IMG   64
#define D_DIM   256
#define T_DIM   4096
#define K_CL    64
#define TT      128                    // pixels per tile
#define S_SPLIT 8                      // workgroups per image
#define CHUNKS  (T_DIM / TT / S_SPLIT) // 4 tiles per workgroup

// LDS row pads (strides in bf16 elements); 264*2=528B, 136*2=272B keep 16B
// alignment and give a 4-bank rotation per row (528/4=132 dw -> +4 banks).
#define XTD_P 264
#define XDT_P 136
#define WL_P  264
#define AL_P  136

// --- WMMA fragment loaders (contiguous k = 16h + s order, same for A and B) ---------------
// B matrix (KxN, 16-bit): lane holds column N=lane%16, 16 *contiguous* K
// values starting at K = (lane<16 ? 0 : 16) within the 32-chunk.
__device__ __forceinline__ v16bf ld16_contig(const __bf16* p) {
    v8bf a = *(const v8bf*)(p);
    v8bf b = *(const v8bf*)(p + 8);
    return __builtin_shufflevector(a, b, 0,1,2,3,4,5,6,7,8,9,10,11,12,13,14,15);
}
// A matrix (MxK, 16-bit): lane holds row M=lane%16, 16 contiguous K values
// starting at K = (lane<16 ? 0 : 16) within the 32-chunk.
__device__ __forceinline__ v16bf ld16_split(const __bf16* p) {
    v8bf a = *(const v8bf*)(p);
    v8bf b = *(const v8bf*)(p + 8);
    return __builtin_shufflevector(a, b, 0,1,2,3,4,5,6,7,8,9,10,11,12,13,14,15);
}

__device__ __forceinline__ v8f wmma_bf16(v16bf a, v16bf b, v8f c) {
    return __builtin_amdgcn_wmma_f32_16x16x32_bf16(
        /*neg_a=*/false, a, /*neg_b=*/false, b,
        /*c_mod=*/(short)0, c, /*reuse_a=*/false, /*reuse_b=*/false);
}

// ---------------------------------------------------------------------------
// Kernel 1: fused normalize + GEMM1 + softmax + GEMM2 partial accumulation
// grid = (S_SPLIT, N_IMG), block = 256 (8 waves)
// ---------------------------------------------------------------------------
__global__ __launch_bounds__(256, 1)
void netvlad_phase1(const float* __restrict__ x,
                    const float* __restrict__ conv_w,
                    float* __restrict__ vlad_ws,   // [N][K][D] fp32, pre-zeroed
                    float* __restrict__ asum_ws)   // [N][K]   fp32, pre-zeroed
{
    __shared__ __align__(16) __bf16 XTD[TT][XTD_P];     // x tile, [t][d]
    __shared__ __align__(16) __bf16 XDT[D_DIM][XDT_P];  // x tile, [d][t]
    __shared__ __align__(16) __bf16 WL [K_CL][WL_P];    // conv_w bf16 [k][d]
    __shared__ __align__(16) __bf16 AL [K_CL][AL_P];    // a' bf16 [k][t]
    __shared__ float SSQ[TT];       // per-pixel sum of squares -> inv norm
    __shared__ float ASUML[K_CL];   // local a_sum accumulator

    const int tid  = threadIdx.x;
    const int lane = tid & 31;
    const int wv   = tid >> 5;        // wave 0..7
    const int lrow = lane & 15;
    const bool hi  = lane >= 16;
    const int n    = blockIdx.y;
    const int s    = blockIdx.x;

    // Stage conv_w (K=64 x D=256) as bf16 once.
    for (int idx = tid; idx < K_CL * D_DIM; idx += 256) {
        int k = idx >> 8, d = idx & 255;
        WL[k][d] = (__bf16)conv_w[idx];
    }
    if (tid < K_CL) ASUML[tid] = 0.f;

    v8f acc[8];
#pragma unroll
    for (int j = 0; j < 8; ++j) acc[j] = {};

    const size_t nbase = (size_t)n * D_DIM * T_DIM;

    for (int c = 0; c < CHUNKS; ++c) {
        const int t0 = (s * CHUNKS + c) * TT;

        __syncthreads();                      // previous chunk fully consumed
        if (tid < TT) SSQ[tid] = 0.f;
        __syncthreads();

        // ---- Phase A: global load (coalesced float4), bf16 to both layouts,
        //      accumulate per-pixel sum of squares.
        {
            const int tq = (tid & 31) * 4;    // 4 consecutive pixels
            float s0 = 0.f, s1 = 0.f, s2 = 0.f, s3 = 0.f;
            const float* xg = x + nbase + t0 + tq;
#pragma unroll 4
            for (int i = 0; i < 32; ++i) {
                const int d = wv + i * 8;     // wave covers d = wv mod 8
                const float4 v = *(const float4*)(xg + (size_t)d * T_DIM);
                const __bf16 b0 = (__bf16)v.x, b1 = (__bf16)v.y;
                const __bf16 b2 = (__bf16)v.z, b3 = (__bf16)v.w;
                *(v4bf*)&XDT[d][tq] = (v4bf){b0, b1, b2, b3};
                XTD[tq + 0][d] = b0; XTD[tq + 1][d] = b1;
                XTD[tq + 2][d] = b2; XTD[tq + 3][d] = b3;
                s0 += v.x * v.x; s1 += v.y * v.y;
                s2 += v.z * v.z; s3 += v.w * v.w;
            }
            atomicAdd(&SSQ[tq + 0], s0); atomicAdd(&SSQ[tq + 1], s1);
            atomicAdd(&SSQ[tq + 2], s2); atomicAdd(&SSQ[tq + 3], s3);
        }
        __syncthreads();
        if (tid < TT) SSQ[tid] = 1.0f / fmaxf(sqrtf(SSQ[tid]), 1e-12f);
        __syncthreads();

        // ---- Phase B: GEMM1 (logits) + softmax + a' to LDS ----------------
        // Wave wv owns pixel tile [wv*16, wv*16+16) x all 4 k-tiles.
        {
            v8f c0 = {}, c1 = {}, c2 = {}, c3 = {};
            const __bf16* xr = &XTD[wv * 16 + lrow][0];
            const int boff = hi ? 16 : 0;
            const int aoff = hi ? 16 : 0;
#pragma unroll
            for (int d0 = 0; d0 < D_DIM; d0 += 32) {
                v16bf bf = ld16_contig(xr + d0 + boff);
                v16bf a0 = ld16_split(&WL[ 0 + lrow][d0 + aoff]);
                v16bf a1 = ld16_split(&WL[16 + lrow][d0 + aoff]);
                v16bf a2 = ld16_split(&WL[32 + lrow][d0 + aoff]);
                v16bf a3 = ld16_split(&WL[48 + lrow][d0 + aoff]);
                c0 = wmma_bf16(a0, bf, c0);
                c1 = wmma_bf16(a1, bf, c1);
                c2 = wmma_bf16(a2, bf, c2);
                c3 = wmma_bf16(a3, bf, c3);
            }
            // logits of normalized x = raw logits * inv[t]
            const float iv = SSQ[wv * 16 + lrow];
            float m = -3.4e38f;
#pragma unroll
            for (int r = 0; r < 8; ++r) {
                c0[r] *= iv; c1[r] *= iv; c2[r] *= iv; c3[r] *= iv;
                m = fmaxf(m, fmaxf(fmaxf(c0[r], c1[r]), fmaxf(c2[r], c3[r])));
            }
            m = fmaxf(m, __shfl_xor(m, 16, 32));   // other 32 k's live in lane^16
            float ssum = 0.f;
#pragma unroll
            for (int r = 0; r < 8; ++r) {
                c0[r] = __expf(c0[r] - m); ssum += c0[r];
                c1[r] = __expf(c1[r] - m); ssum += c1[r];
                c2[r] = __expf(c2[r] - m); ssum += c2[r];
                c3[r] = __expf(c3[r] - m); ssum += c3[r];
            }
            ssum += __shfl_xor(ssum, 16, 32);
            const float rs = 1.0f / ssum;
            const int tcol = wv * 16 + lrow;
            const int kof  = hi ? 8 : 0;
#pragma unroll
            for (int r = 0; r < 8; ++r) {
                const float a0v = c0[r] * rs, a1v = c1[r] * rs;
                const float a2v = c2[r] * rs, a3v = c3[r] * rs;
                // a' = a * inv[t] so that a' @ x_raw == a @ x_normalized
                AL[ 0 + kof + r][tcol] = (__bf16)(a0v * iv);
                AL[16 + kof + r][tcol] = (__bf16)(a1v * iv);
                AL[32 + kof + r][tcol] = (__bf16)(a2v * iv);
                AL[48 + kof + r][tcol] = (__bf16)(a3v * iv);
                c0[r] = a0v; c1[r] = a1v; c2[r] = a2v; c3[r] = a3v;
            }
            // a_sum over this wave's 16 pixels (xor-reduce within each half)
#pragma unroll
            for (int off = 1; off < 16; off <<= 1) {
#pragma unroll
                for (int r = 0; r < 8; ++r) {
                    c0[r] += __shfl_xor(c0[r], off, 32);
                    c1[r] += __shfl_xor(c1[r], off, 32);
                    c2[r] += __shfl_xor(c2[r], off, 32);
                    c3[r] += __shfl_xor(c3[r], off, 32);
                }
            }
            if (lrow == 0) {
#pragma unroll
                for (int r = 0; r < 8; ++r) {
                    atomicAdd(&ASUML[ 0 + kof + r], c0[r]);
                    atomicAdd(&ASUML[16 + kof + r], c1[r]);
                    atomicAdd(&ASUML[32 + kof + r], c2[r]);
                    atomicAdd(&ASUML[48 + kof + r], c3[r]);
                }
            }
        }
        __syncthreads();

        // ---- Phase C: GEMM2 vlad += a'(K x 128t) @ x^T(128t x D) ----------
        // Wave wv: k-tile = wv/2; d-columns = (wv&1)*128 .. +128.
        {
            const int mt2 = wv >> 1;
            const __bf16* ar = &AL[mt2 * 16 + lrow][0];
            const int aoff = hi ? 16 : 0;
            const int boff = hi ? 16 : 0;
            v16bf af0 = ld16_split(ar +  0 + aoff);
            v16bf af1 = ld16_split(ar + 32 + aoff);
            v16bf af2 = ld16_split(ar + 64 + aoff);
            v16bf af3 = ld16_split(ar + 96 + aoff);
#pragma unroll
            for (int j = 0; j < 8; ++j) {
                const __bf16* xr2 = &XDT[((wv & 1) * 8 + j) * 16 + lrow][0];
                v16bf b0 = ld16_contig(xr2 +  0 + boff);
                acc[j] = wmma_bf16(af0, b0, acc[j]);
                v16bf b1 = ld16_contig(xr2 + 32 + boff);
                acc[j] = wmma_bf16(af1, b1, acc[j]);
                v16bf b2 = ld16_contig(xr2 + 64 + boff);
                acc[j] = wmma_bf16(af2, b2, acc[j]);
                v16bf b3 = ld16_contig(xr2 + 96 + boff);
                acc[j] = wmma_bf16(af3, b3, acc[j]);
            }
        }
    } // chunks

    // ---- Flush partial vlad and a_sum to global workspace (L2-resident) ---
    {
        const int mt2 = wv >> 1;
        const int kb  = mt2 * 16 + (hi ? 8 : 0);
#pragma unroll
        for (int j = 0; j < 8; ++j) {
            const int d = ((wv & 1) * 8 + j) * 16 + lrow;
            float* dst = &vlad_ws[((size_t)n * K_CL + kb) * D_DIM + d];
#pragma unroll
            for (int r = 0; r < 8; ++r)
                atomicAdd(dst + (size_t)r * D_DIM, acc[j][r]);
        }
    }
    __syncthreads();
    if (tid < K_CL) atomicAdd(&asum_ws[n * K_CL + tid], ASUML[tid]);
}

// ---------------------------------------------------------------------------
// Kernel 2: per (n,k) row: vlad - a_sum*centroid, intra L2-normalize over D,
// accumulate per-image sum of squares of the normalized rows.
// ---------------------------------------------------------------------------
__global__ __launch_bounds__(256)
void netvlad_finalize_rows(const float* __restrict__ vlad_ws,
                           const float* __restrict__ asum_ws,
                           const float* __restrict__ centroids,
                           float* __restrict__ out,
                           float* __restrict__ nss_ws)
{
    __shared__ float red[8];
    const int row = blockIdx.x;        // row = n*64 + k
    const int k   = row & (K_CL - 1);
    const int n   = row >> 6;
    const int d   = threadIdx.x;

    float y = vlad_ws[(size_t)row * D_DIM + d]
            - asum_ws[row] * centroids[(size_t)k * D_DIM + d];
    float ss = y * y;
#pragma unroll
    for (int off = 1; off < 32; off <<= 1) ss += __shfl_xor(ss, off, 32);
    const int lane = d & 31, wv = d >> 5;
    if (lane == 0) red[wv] = ss;
    __syncthreads();
    float tot = 0.f;
#pragma unroll
    for (int i = 0; i < 8; ++i) tot += red[i];
    const float invr = 1.0f / fmaxf(sqrtf(tot), 1e-12f);
    out[(size_t)row * D_DIM + d] = y * invr;
    if (d == 0) atomicAdd(&nss_ws[n], tot * invr * invr);
}

// ---------------------------------------------------------------------------
// Kernel 3: final per-image L2 scale over the flattened K*D vector.
// ---------------------------------------------------------------------------
__global__ __launch_bounds__(256)
void netvlad_scale(float* __restrict__ out, const float* __restrict__ nss_ws)
{
    const int n = blockIdx.x;
    const float sc = 1.0f / fmaxf(sqrtf(nss_ws[n]), 1e-12f);
    const size_t base = (size_t)n * K_CL * D_DIM;
    for (int i = threadIdx.x; i < K_CL * D_DIM; i += 256)
        out[base + i] *= sc;
}

// ---------------------------------------------------------------------------
extern "C" void kernel_launch(void* const* d_in, const int* in_sizes, int n_in,
                              void* d_out, int out_size, void* d_ws, size_t ws_size,
                              hipStream_t stream)
{
    (void)in_sizes; (void)n_in; (void)out_size; (void)ws_size;
    const float* x         = (const float*)d_in[0];   // [N,D,H,W] fp32
    const float* conv_w    = (const float*)d_in[1];   // [K,D] fp32
    const float* centroids = (const float*)d_in[2];   // [K,D] fp32
    float* out = (float*)d_out;                       // [N, K*D] fp32

    float* vlad_ws = (float*)d_ws;                              // 4 MB
    float* asum_ws = vlad_ws + (size_t)N_IMG * K_CL * D_DIM;    // 16 KB
    float* nss_ws  = asum_ws + N_IMG * K_CL;                    // 256 B

    const size_t zero_bytes =
        ((size_t)N_IMG * K_CL * D_DIM + (size_t)N_IMG * K_CL + N_IMG) * sizeof(float);
    hipMemsetAsync(d_ws, 0, zero_bytes, stream);

    netvlad_phase1<<<dim3(S_SPLIT, N_IMG), 256, 0, stream>>>(x, conv_w, vlad_ws, asum_ws);
    netvlad_finalize_rows<<<N_IMG * K_CL, 256, 0, stream>>>(vlad_ws, asum_ws, centroids,
                                                            out, nss_ws);
    netvlad_scale<<<N_IMG, 256, 0, stream>>>(out, nss_ws);
}
